// SAM_Module_37434934952403
// MI455X (gfx1250) — matrix operand order repaired and checked by reading, not compiled, not run on hardware
//
#include <hip/hip_runtime.h>
#include <hip/hip_bf16.h>

typedef __attribute__((ext_vector_type(16))) __bf16 v16bf;
typedef __attribute__((ext_vector_type(8)))  __bf16 v8bf;
typedef __attribute__((ext_vector_type(8)))  float  v8f;

#define HW 4096   // h*w = 64*64
#define CC 128    // C*R*R = 32*4
#define NB 4      // batch
#define BN 64     // j-block width
#define NKS 4     // K-steps of 32 over CC=128
#define WAVES 4   // waves per block

// ---------------------------------------------------------------------------
// Kernel 1: de-pixel-shuffle + cosine normalization.
//   qn[b][i][c] = bf16( q[b][c][i] / ||q[b][:][i]|| )   (row-major over c)
//   v [b][c][i] = bf16( q[b][c][i] )                    (row-major over i)
// ---------------------------------------------------------------------------
__global__ void shuffle_norm_kernel(const float* __restrict__ x,
                                    __bf16* __restrict__ qn,
                                    __bf16* __restrict__ vv)
{
    int idx = blockIdx.x * blockDim.x + threadIdx.x;   // 0 .. NB*HW-1
    int b = idx >> 12;
    int i = idx & (HW - 1);
    int hy = i >> 6, wx = i & 63;
    const float* xb = x + (size_t)b * 32 * 128 * 128;

    float sumsq = 0.f;
    #pragma unroll 4
    for (int c = 0; c < CC; ++c) {
        int co = c >> 2, ry = (c >> 1) & 1, rx = c & 1;
        float val = xb[((size_t)co * 128 + (hy * 2 + ry)) * 128 + (wx * 2 + rx)];
        sumsq += val * val;
        vv[((size_t)(b * CC + c)) * HW + i] = (__bf16)val;
    }
    float inv_d = rsqrtf(sumsq);
    __bf16* qrow = qn + ((size_t)(b * HW + i)) * CC;
    #pragma unroll 4
    for (int c = 0; c < CC; ++c) {
        int co = c >> 2, ry = (c >> 1) & 1, rx = c & 1;
        float val = xb[((size_t)co * 128 + (hy * 2 + ry)) * 128 + (wx * 2 + rx)];
        qrow[c] = (__bf16)(val * inv_d);
    }
}

// ---------------------------------------------------------------------------
// Kernel 2: fused flash-attention over positions (seq=4096, head dim=128),
// Q=K=qn (cosine scores in [-1,1]), V=q.
// Fixed-shift softmax: attn = exp(S-1)/rowsum(exp(S-1)) == softmax(S).
// Fragment loads are explicitly double-buffered (software pipeline) so the
// register allocator cannot coalesce them and DS loads overlap the XDL pipe.
// ---------------------------------------------------------------------------
__launch_bounds__(128, 1)
__global__ void attn_kernel(const float* __restrict__ x,
                            const __bf16* __restrict__ qn,
                            const __bf16* __restrict__ vv,
                            const float* __restrict__ gamma_p,
                            float* __restrict__ out)
{
    __shared__ __bf16 sQ[BN][CC];          // key tile 64x128,   16 KB
    __shared__ __bf16 sV[CC][BN];          // value tile [c][j], 16 KB
    __shared__ __bf16 sP[WAVES][16][BN];   // per-wave P relayout scratch, 8 KB

    const int tid  = threadIdx.x;
    const int wave = tid >> 5;
    const int lane = tid & 31;
    const int lg   = lane >> 4;            // lane group (0: lanes 0-15, 1: 16-31)
    const int ln   = lane & 15;

    const int b  = blockIdx.x >> 6;        // 64 row-blocks per batch
    const int ib = blockIdx.x & 63;
    const int I0 = ib * 64 + wave * 16;    // this wave's first row (i)

    const float gamma = gamma_p[0];
    const __bf16* qnb = qn + (size_t)b * HW * CC;
    const __bf16* vb  = vv + (size_t)b * CC * HW;

    // --- Preload A-fragments for this wave's 16 query rows (K = 128) ---
    // A 16x32 bf16 layout: M = lane%16; lanes 0-15 hold K = ks*32+{0..15},
    // lanes 16-31 hold K = ks*32+{16..31}.
    v16bf afrag[NKS];
    {
        const __bf16* qbase = qnb + (size_t)(I0 + ln) * CC;
        #pragma unroll
        for (int ks = 0; ks < NKS; ++ks) {
            union { v16bf w; v8bf h[2]; } u;
            u.h[0] = *(const v8bf*)(qbase + ks * 32 + lg * 16);
            u.h[1] = *(const v8bf*)(qbase + ks * 32 + lg * 16 + 8);
            afrag[ks] = u.w;
        }
    }

    v8f   O[8];                            // 16x128 output accum (8 N-tiles)
    float lsum[8];                         // per-lane partial row sums
    #pragma unroll
    for (int t = 0; t < 8; ++t) O[t] = (v8f){};
    #pragma unroll
    for (int r = 0; r < 8; ++r) lsum[r] = 0.f;

    for (int j0 = 0; j0 < HW; j0 += BN) {
        __syncthreads();                   // previous iter done reading LDS
        // cooperative load of key tile: 64 rows x 128 bf16 (512 x 32B chunks)
        #pragma unroll
        for (int k = 0; k < 4; ++k) {
            int idx = tid + k * 128;
            int row = idx >> 3, chunk = idx & 7;
            *(v16bf*)&sQ[row][chunk * 16] =
                *(const v16bf*)(qnb + (size_t)(j0 + row) * CC + chunk * 16);
        }
        // cooperative load of value tile transposed: 128 rows x 64 bf16
        #pragma unroll
        for (int k = 0; k < 4; ++k) {
            int idx = tid + k * 128;
            int row = idx >> 2, quar = idx & 3;
            *(v16bf*)&sV[row][quar * 16] =
                *(const v16bf*)(vb + (size_t)row * HW + j0 + quar * 16);
        }
        __syncthreads();

        // --- scores: S(16 x 64) = Qn_I . Qn_J^T, pipelined over K-steps ---
        // B 32x16 layout: N = lane%16, lanes 0-15 K=+0..15, 16-31 K=+16..31.
        v8f S[4];
        #pragma unroll
        for (int jt = 0; jt < 4; ++jt) S[jt] = (v8f){};

        v16bf bfr[2][4];                   // double-buffered B-fragments
        #pragma unroll
        for (int jt = 0; jt < 4; ++jt)
            bfr[0][jt] = *(const v16bf*)&sQ[jt * 16 + ln][lg * 16];
        #pragma unroll
        for (int ks = 0; ks < NKS; ++ks) {
            const int cur = ks & 1;
            if (ks + 1 < NKS) {            // prefetch next K-step's fragments
                #pragma unroll
                for (int jt = 0; jt < 4; ++jt)
                    bfr[cur ^ 1][jt] =
                        *(const v16bf*)&sQ[jt * 16 + ln][(ks + 1) * 32 + lg * 16];
            }
            #pragma unroll
            for (int jt = 0; jt < 4; ++jt)
                S[jt] = __builtin_amdgcn_wmma_f32_16x16x32_bf16(
                            false, afrag[ks], false, bfr[cur][jt], (short)0,
                            S[jt], false, false);
        }

        // --- p = exp(S - 1) (bounded, no max needed); stage P, sum rows ---
        // C/D layout: row M = r + 8*lg, col N = jt*16 + ln.
        #pragma unroll
        for (int jt = 0; jt < 4; ++jt) {
            #pragma unroll
            for (int r = 0; r < 8; ++r) {
                float p = __expf(S[jt][r] - 1.0f);
                lsum[r] += p;
                sP[wave][r + lg * 8][jt * 16 + ln] = (__bf16)p;
            }
        }
        // lanes of a wave run in lockstep; just drain LDS stores
        asm volatile("s_wait_dscnt 0x0" ::: "memory");

        // P A-fragments (16 x 64 -> 2 K-steps of 32)
        v16bf pfrag[2];
        #pragma unroll
        for (int ks2 = 0; ks2 < 2; ++ks2) {
            union { v16bf w; v8bf h[2]; } up;
            up.h[0] = *(const v8bf*)&sP[wave][ln][ks2 * 32 + lg * 16];
            up.h[1] = *(const v8bf*)&sP[wave][ln][ks2 * 32 + lg * 16 + 8];
            pfrag[ks2] = up.w;
        }

        // --- O(16x128) += P(16x64) . V(64x128), pipelined over 4 steps ---
        // step s: ks2 = s>>1, tile group tg = s&1 (4 O-tiles each).
        v16bf vf[2][4];                    // double-buffered V-fragments
        #pragma unroll
        for (int u = 0; u < 4; ++u)
            vf[0][u] = *(const v16bf*)&sV[u * 16 + ln][lg * 16];
        #pragma unroll
        for (int s = 0; s < 4; ++s) {
            const int cur = s & 1;
            if (s + 1 < 4) {               // prefetch next step's fragments
                const int ks2n = (s + 1) >> 1, tgn = (s + 1) & 1;
                #pragma unroll
                for (int u = 0; u < 4; ++u)
                    vf[cur ^ 1][u] = *(const v16bf*)
                        &sV[(tgn * 4 + u) * 16 + ln][ks2n * 32 + lg * 16];
            }
            const int ks2 = s >> 1, tg = s & 1;
            #pragma unroll
            for (int u = 0; u < 4; ++u)
                O[tg * 4 + u] = __builtin_amdgcn_wmma_f32_16x16x32_bf16(
                                    false, pfrag[ks2], false, vf[cur][u], (short)0,
                                    O[tg * 4 + u], false, false);
        }
    }

    // --- one-time row-sum reduction across the 16 N-lanes ---
    float inv_l[8];
    #pragma unroll
    for (int r = 0; r < 8; ++r) {
        float s = lsum[r];
        #pragma unroll
        for (int off = 8; off >= 1; off >>= 1)
            s += __shfl_xor(s, off, 32);
        inv_l[r] = 1.f / s;
    }

    // --- epilogue: out = gamma * O/l + xs, pixel-shuffled back to (B,C,H,W) ---
    const float* xb = x   + (size_t)b * 32 * 128 * 128;
    float*       ob = out + (size_t)b * 32 * 128 * 128;
    #pragma unroll
    for (int t = 0; t < 8; ++t) {
        int c  = t * 16 + ln;
        int co = c >> 2, ry = (c >> 1) & 1, rx = c & 1;
        #pragma unroll
        for (int r = 0; r < 8; ++r) {
            int i  = I0 + r + lg * 8;
            int hy = i >> 6, wx = i & 63;
            size_t off = ((size_t)co * 128 + hy * 2 + ry) * 128 + wx * 2 + rx;
            ob[off] = gamma * (O[t][r] * inv_l[r]) + xb[off];
        }
    }
}

// ---------------------------------------------------------------------------
extern "C" void kernel_launch(void* const* d_in, const int* in_sizes, int n_in,
                              void* d_out, int out_size, void* d_ws, size_t ws_size,
                              hipStream_t stream) {
    const float* x     = (const float*)d_in[0];
    const float* gamma = (const float*)d_in[1];
    float*       out   = (float*)d_out;

    // workspace: qn (bf16, [B][hw][C]) then v (bf16, [B][C][hw]) — 4 MB each
    __bf16* qn = (__bf16*)d_ws;
    __bf16* vv = (__bf16*)((char*)d_ws + (size_t)NB * HW * CC * sizeof(__bf16));

    shuffle_norm_kernel<<<(NB * HW) / 256, 256, 0, stream>>>(x, qn, vv);
    attn_kernel<<<NB * (HW / BN), 128, 0, stream>>>(x, qn, vv, gamma, out);
}
